// SynthesisConstBlock_62414464745789
// MI455X (gfx1250) — matrix operand order repaired and checked by reading, not compiled, not run on hardware
//
#include <hip/hip_runtime.h>
#include <math.h>

// ---------------------------------------------------------------------------
// StyleGAN2 SynthesisConstBlock, restructured as 3 bf16 WMMA GEMMs on gfx1250.
//   s  = w0 @ (mod_w/sqrt(512)) + mod_b + 1            [1024 x 512]
//   T  = conv(const, w) folded per (ci -> co*16+pq)    [512  x 8192] (precompute)
//   y  = (s @ T) * rsqrt(s^2 @ wsq + eps) + noise/bias -> lrelu*sqrt(2)
// Each wave computes a 32(M) x 64(N) tile: 8 WMMA accumulators, 2 A fragments
// reused across 4 B fragments per K-step (halves B-side L2 traffic vs 16x64).
// ---------------------------------------------------------------------------

typedef __attribute__((ext_vector_type(16))) __bf16 v16bf;
typedef __attribute__((ext_vector_type(8)))  float  v8f;

#define DEV static __device__ __forceinline__

constexpr int BATCH = 1024;
constexpr int LAT   = 512;     // latent
constexpr int FM    = 512;     // fmaps (cin == cout)
constexpr int RES2  = 16;      // 4x4
constexpr int NY    = FM * RES2;  // 8192 output columns of main GEMM

constexpr float COEF_W   = 0.014731391f;   // 1/sqrt(3*3*512)
constexpr float COEF_MOD = 0.044194174f;   // 1/sqrt(512)
constexpr float LRELU_A  = 0.2f;
constexpr float ACT_GAIN = 1.41421356237f; // sqrt(2)
constexpr float DEMOD_EPS = 1e-8f;

// f32 -> bf16 (round to nearest even), kept as raw u16 for storage.
DEV unsigned short f2bf(float f) {
  unsigned u = __float_as_uint(f);
  unsigned r = (u + 0x7FFFu + ((u >> 16) & 1u)) >> 16;
  return (unsigned short)r;
}

union Frag16 { v16bf v; uint4 u[2]; };

// ---------------------------------------------------------------------------
// Core WMMA tile: one wave computes a 32(M) x 64(N) tile, acc[2][4] of 16x16.
// A: [M][K] bf16 row-major.  Bt: [N][K] bf16 row-major (i.e., B transposed).
// Per ISA 7.12.2 (wave32):
//   A frag lane L: m = L%16, elems e0..7  -> K = k0 + 8*(L>=16) + e,
//                  elems e8..15 -> K = k0 + 16 + 8*(L>=16) + (e-8).
//   B frag lane L: n = L%16, same K mapping as A (two runs of 8).
// Both are contiguous runs in K -> b128 loads.
// ---------------------------------------------------------------------------
DEV void mma32x64(const unsigned short* __restrict__ A,
                  const unsigned short* __restrict__ Bt,
                  int K, int m0, int n0, v8f acc[2][4]) {
  const int lane = (int)(threadIdx.x & 31u);
  const int hi16 = lane >> 4;     // 0 for lanes 0-15, 1 for lanes 16-31
  const int lm   = lane & 15;
  const unsigned short* arow0 = A + (size_t)(m0 + lm) * K;
  const unsigned short* arow1 = arow0 + (size_t)16 * K;
  for (int k0 = 0; k0 < K; k0 += 32) {
    Frag16 af[2];
    af[0].u[0] = *reinterpret_cast<const uint4*>(arow0 + k0 + hi16 * 8);
    af[0].u[1] = *reinterpret_cast<const uint4*>(arow0 + k0 + hi16 * 8 + 16);
    af[1].u[0] = *reinterpret_cast<const uint4*>(arow1 + k0 + hi16 * 8);
    af[1].u[1] = *reinterpret_cast<const uint4*>(arow1 + k0 + hi16 * 8 + 16);
#pragma unroll
    for (int t = 0; t < 4; ++t) {
      Frag16 bfr;
      const unsigned short* brow =
          Bt + (size_t)(n0 + t * 16 + lm) * K + k0 + hi16 * 8;
      const uint4* bp = reinterpret_cast<const uint4*>(brow);
      bfr.u[0] = bp[0];
      bfr.u[1] = bp[2];
      acc[0][t] = __builtin_amdgcn_wmma_f32_16x16x32_bf16(
          false, af[0].v, false, bfr.v, (short)0, acc[0][t], false, false);
      acc[1][t] = __builtin_amdgcn_wmma_f32_16x16x32_bf16(
          false, af[1].v, false, bfr.v, (short)0, acc[1][t], false, false);
    }
  }
}

DEV void zero_acc(v8f acc[2][4]) {
#pragma unroll
  for (int s = 0; s < 2; ++s)
#pragma unroll
    for (int t = 0; t < 4; ++t)
#pragma unroll
      for (int i = 0; i < 8; ++i) acc[s][t][i] = 0.0f;
}

// ---------------------------------------------------------------------------
// Conversion kernels (one-time, tiny)
// ---------------------------------------------------------------------------
__global__ void k_cvt_w0(const float* __restrict__ w0,
                         unsigned short* __restrict__ w0_bf, int n) {
  int i = blockIdx.x * 256 + threadIdx.x;
  if (i < n) w0_bf[i] = f2bf(w0[i]);
}

// mod_w is [latent][fmaps]; produce Bt layout [fmaps][latent] with 1/sqrt(L).
__global__ void k_cvt_modw(const float* __restrict__ mod_w,
                           unsigned short* __restrict__ wm_bt) {
  int i = blockIdx.x * 256 + threadIdx.x;
  if (i < LAT * FM) {
    int l = i / FM, c = i % FM;
    wm_bt[(size_t)c * LAT + l] = f2bf(mod_w[i] * COEF_MOD);
  }
}

// ---------------------------------------------------------------------------
// Precompute T_bt[n = co*16+pq][ci]  (bf16, [8192][512])  and
//            wsq_bt[co][ci]          (bf16, [512][512]).
// grid (FM/64, FM/16): blockIdx.x = ci tile of 64, blockIdx.y = co tile of 16.
// LDS transpose so global stores are full 128B rows.
// ---------------------------------------------------------------------------
__global__ void k_prep_T(const float* __restrict__ conv_w,   // [3][3][cin][cout]
                         const float* __restrict__ cst,      // [cin][4][4]
                         unsigned short* __restrict__ T_bt,
                         unsigned short* __restrict__ wsq_bt) {
  __shared__ __align__(16) unsigned short ldsT[256 * 64];  // [n_local][ci_local]
  __shared__ __align__(16) unsigned short ldsW[16 * 64];   // [co_local][ci_local]
  const int ci_t = blockIdx.x;  // 0..7
  const int co_t = blockIdx.y;  // 0..31
#pragma unroll
  for (int r = 0; r < 4; ++r) {
    int p = (int)threadIdx.x + r * 256;   // 0..1023
    int ci_l = p >> 4;                    // 0..63
    int co_l = p & 15;                    // 0..15
    int ci = ci_t * 64 + ci_l;
    int co = co_t * 16 + co_l;
    float w9[9];
    float ws = 0.0f;
#pragma unroll
    for (int t = 0; t < 9; ++t) {
      float w = conv_w[(size_t)t * FM * FM + (size_t)ci * FM + co] * COEF_W;
      w9[t] = w;
      ws += w * w;
    }
    ldsW[co_l * 64 + ci_l] = f2bf(ws);
#pragma unroll
    for (int pq = 0; pq < 16; ++pq) {
      int y = pq >> 2, x = pq & 3;
      float a = 0.0f;
#pragma unroll
      for (int kh = 0; kh < 3; ++kh)
#pragma unroll
        for (int kw = 0; kw < 3; ++kw) {
          int yy = y + kh - 1, xx = x + kw - 1;
          if (yy >= 0 && yy < 4 && xx >= 0 && xx < 4)
            a += cst[ci * 16 + yy * 4 + xx] * w9[kh * 3 + kw];
        }
      ldsT[(co_l * 16 + pq) * 64 + ci_l] = f2bf(a);
    }
  }
  __syncthreads();
  {  // 128B-per-thread coalesced writeout of T
    int n_l = (int)threadIdx.x;                 // 0..255
    int n = co_t * 256 + n_l;                   // co*16+pq
    const uint4* src = reinterpret_cast<const uint4*>(&ldsT[n_l * 64]);
    uint4* dst = reinterpret_cast<uint4*>(&T_bt[(size_t)n * FM + ci_t * 64]);
#pragma unroll
    for (int i = 0; i < 8; ++i) dst[i] = src[i];
  }
  if (threadIdx.x < 16) {
    int co = co_t * 16 + (int)threadIdx.x;
    const uint4* src = reinterpret_cast<const uint4*>(&ldsW[threadIdx.x * 64]);
    uint4* dst = reinterpret_cast<uint4*>(&wsq_bt[(size_t)co * FM + ci_t * 64]);
#pragma unroll
    for (int i = 0; i < 8; ++i) dst[i] = src[i];
  }
}

// ---------------------------------------------------------------------------
// GEMM 1: s = w0 @ modw' + mod_b + 1  -> store s (bf16) and s^2 (bf16)
// ---------------------------------------------------------------------------
__global__ void k_gemm_s(const unsigned short* __restrict__ A,   // w0 bf16
                         const unsigned short* __restrict__ Bt,  // wm_bt
                         const float* __restrict__ mod_b,
                         unsigned short* __restrict__ s_bf,
                         unsigned short* __restrict__ s2_bf) {
  int wid = (int)((blockIdx.x * 256 + threadIdx.x) >> 5);
  const int NT = FM / 64;
  int m0 = (wid / NT) * 32, n0 = (wid % NT) * 64;
  v8f acc[2][4];
  zero_acc(acc);
  mma32x64(A, Bt, LAT, m0, n0, acc);
  const int lane = (int)(threadIdx.x & 31u);
  const int lm = lane & 15, mo = (lane >> 4) * 8;
#pragma unroll
  for (int s = 0; s < 2; ++s)
#pragma unroll
    for (int t = 0; t < 4; ++t) {
      int n = n0 + t * 16 + lm;
      float b = mod_b[n] + 1.0f;
#pragma unroll
      for (int r = 0; r < 8; ++r) {
        int m = m0 + s * 16 + mo + r;
        float v = acc[s][t][r] + b;
        s_bf[(size_t)m * FM + n] = f2bf(v);
        s2_bf[(size_t)m * FM + n] = f2bf(v * v);
      }
    }
}

// ---------------------------------------------------------------------------
// GEMM 2: d = rsqrt(s^2 @ wsq + eps)   [1024 x 512] f32
// ---------------------------------------------------------------------------
__global__ void k_gemm_d(const unsigned short* __restrict__ A,   // s2 bf16
                         const unsigned short* __restrict__ Bt,  // wsq_bt
                         float* __restrict__ d) {
  int wid = (int)((blockIdx.x * 256 + threadIdx.x) >> 5);
  const int NT = FM / 64;
  int m0 = (wid / NT) * 32, n0 = (wid % NT) * 64;
  v8f acc[2][4];
  zero_acc(acc);
  mma32x64(A, Bt, FM, m0, n0, acc);
  const int lane = (int)(threadIdx.x & 31u);
  const int lm = lane & 15, mo = (lane >> 4) * 8;
#pragma unroll
  for (int s = 0; s < 2; ++s)
#pragma unroll
    for (int t = 0; t < 4; ++t) {
      int n = n0 + t * 16 + lm;
#pragma unroll
      for (int r = 0; r < 8; ++r) {
        int m = m0 + s * 16 + mo + r;
        d[(size_t)m * FM + n] = rsqrtf(acc[s][t][r] + DEMOD_EPS);
      }
    }
}

// ---------------------------------------------------------------------------
// GEMM 3: y = s @ T, then demod + noise + bias + lrelu*sqrt(2) -> d_out (f32)
// Output layout [B][cout][4][4] flat: m*8192 + n  with n = co*16 + pq.
// ---------------------------------------------------------------------------
__global__ void k_gemm_y(const unsigned short* __restrict__ A,   // s bf16
                         const unsigned short* __restrict__ Bt,  // T_bt
                         const float* __restrict__ d,
                         const float* __restrict__ noise,        // [B][16]
                         const float* __restrict__ ns,           // [1]
                         const float* __restrict__ bias,         // [cout]
                         float* __restrict__ out) {
  int wid = (int)((blockIdx.x * 256 + threadIdx.x) >> 5);
  const int NT = NY / 64;  // 128
  int m0 = (wid / NT) * 32, n0 = (wid % NT) * 64;
  v8f acc[2][4];
  zero_acc(acc);
  mma32x64(A, Bt, FM, m0, n0, acc);
  const int lane = (int)(threadIdx.x & 31u);
  const int lm = lane & 15, mo = (lane >> 4) * 8;
  const float nstr = ns[0];
#pragma unroll
  for (int s = 0; s < 2; ++s)
#pragma unroll
    for (int t = 0; t < 4; ++t) {
      int n = n0 + t * 16 + lm;
      int co = n >> 4, pq = n & 15;
      float bs = bias[co];
#pragma unroll
      for (int r = 0; r < 8; ++r) {
        int m = m0 + s * 16 + mo + r;
        float v = acc[s][t][r] * d[(size_t)m * FM + co]
                + nstr * noise[(size_t)m * RES2 + pq] + bs;
        v = (v > 0.0f ? v : LRELU_A * v) * ACT_GAIN;
        out[(size_t)m * NY + n] = v;
      }
    }
}

// ---------------------------------------------------------------------------
extern "C" void kernel_launch(void* const* d_in, const int* in_sizes, int n_in,
                              void* d_out, int out_size, void* d_ws, size_t ws_size,
                              hipStream_t stream) {
  const float* w0     = (const float*)d_in[0];
  const float* cst    = (const float*)d_in[1];
  const float* conv_w = (const float*)d_in[2];
  const float* mod_w  = (const float*)d_in[3];
  const float* mod_b  = (const float*)d_in[4];
  const float* noise  = (const float*)d_in[5];
  const float* nstr   = (const float*)d_in[6];
  const float* bias   = (const float*)d_in[7];
  float* out = (float*)d_out;

  char* ws = (char*)d_ws;
  size_t o = 0;
  auto take = [&](size_t bytes) { char* p = ws + o; o = (o + bytes + 255) & ~(size_t)255; return p; };
  unsigned short* w0_bf  = (unsigned short*)take((size_t)BATCH * LAT * 2);
  unsigned short* wm_bt  = (unsigned short*)take((size_t)FM * LAT * 2);
  unsigned short* s_bf   = (unsigned short*)take((size_t)BATCH * FM * 2);
  unsigned short* s2_bf  = (unsigned short*)take((size_t)BATCH * FM * 2);
  unsigned short* T_bt   = (unsigned short*)take((size_t)NY * FM * 2);
  unsigned short* wsq_bt = (unsigned short*)take((size_t)FM * FM * 2);
  float*          d_dem  = (float*)take((size_t)BATCH * FM * 4);
  (void)ws_size; (void)in_sizes; (void)n_in; (void)out_size;

  k_cvt_w0<<<(BATCH * LAT + 255) / 256, 256, 0, stream>>>(w0, w0_bf, BATCH * LAT);
  k_cvt_modw<<<(LAT * FM + 255) / 256, 256, 0, stream>>>(mod_w, wm_bt);
  k_prep_T<<<dim3(FM / 64, FM / 16), 256, 0, stream>>>(conv_w, cst, T_bt, wsq_bt);

  // 32x64 tile per wave, 8 waves per block
  int waves_s = (BATCH / 32) * (FM / 64);        // 256
  k_gemm_s<<<waves_s / 8, 256, 0, stream>>>(w0_bf, wm_bt, mod_b, s_bf, s2_bf);
  k_gemm_d<<<waves_s / 8, 256, 0, stream>>>(s2_bf, wsq_bt, d_dem);
  int waves_y = (BATCH / 32) * (NY / 64);        // 4096
  k_gemm_y<<<waves_y / 8, 256, 0, stream>>>(s_bf, T_bt, d_dem, noise, nstr, bias, out);
}
